// Scan2latent_27212912788018
// MI455X (gfx1250) — compile-verified, matrix operand order checked by reading, not run on hardware
//
#include <hip/hip_runtime.h>
#include <hip/hip_bf16.h>

// ---------------- problem constants ----------------
#define BATCH   32
#define NIN     2048
#define NLAT    512
#define DIMD    512
#define FFD     2048          // MULT*DIM
#define FEATN   51            // sin(24) + cos(24) + xyz(3)
#define LN_EPS  1e-5f

#define LDA_M   72            // padded LDS row stride (elems), main GEMM (K-step 64)
#define LDA_G   40            // padded LDS row stride (elems), GEGLU (K-step 32)

// ---------------- CDNA5 WMMA types ----------------
typedef __attribute__((ext_vector_type(16))) __bf16 v16bf;
typedef __attribute__((ext_vector_type(8)))  float  v8f;
typedef unsigned int u32x4 __attribute__((ext_vector_type(4)));

union FragAB { v16bf v; u32x4 q[2]; };

__device__ __forceinline__ unsigned short f2bf(float x) {
  unsigned int u = __float_as_uint(x);
  u += 0x7FFFu + ((u >> 16) & 1u);          // round-to-nearest-even
  return (unsigned short)(u >> 16);
}
__device__ __forceinline__ float bf2f(unsigned short h) {
  return __uint_as_float(((unsigned int)h) << 16);
}

// gfx1250 async global->LDS copy, GVS mode (SGPR base + 32-bit VGPR offset).
// Tracked by ASYNCcnt; drained with s_wait_asynccnt.
__device__ __forceinline__ void async_b128(unsigned lds_addr, const void* sbase,
                                           unsigned byte_off) {
  asm volatile("global_load_async_to_lds_b128 %0, %1, %2 offset:0"
               :: "v"(lds_addr), "v"(byte_off), "s"(sbase)
               : "memory");
}
#define ASYNC_WAIT(n) asm volatile("s_wait_asynccnt " #n ::: "memory")

// 16x32 bf16 fragment per ISA 7.12.2: lane = (half<<4)|m ; holds
// K = half*8+[0..7] and 16+half*8+[0..7]: two 16-byte chunks at +0 and +16.
__device__ __forceinline__ FragAB lds_frag(const unsigned short* p) {
  FragAB f;
  f.q[0] = *(const u32x4*)(p);
  f.q[1] = *(const u32x4*)(p + 16);
  return f;
}

#define WMMA_BF16(A_, B_, C_) \
  __builtin_amdgcn_wmma_f32_16x16x32_bf16(false, (A_), false, (B_), (short)0, (C_), false, false)

// ---------------- weight prep: f32 [K,N] -> bf16 [N,K] ----------------
__global__ void __launch_bounds__(256)
transpose_to_bf16(const float* __restrict__ W, unsigned short* __restrict__ WT,
                  int K, int N) {
  long i = (long)blockIdx.x * 256 + threadIdx.x;
  long total = (long)K * N;
  if (i < total) {
    long k = i / N, n = i % N;
    WT[n * (long)K + k] = f2bf(W[i]);
  }
}

// ---------------- farthest point sampling ----------------
__global__ void __launch_bounds__(256)
fps_kernel(const float* __restrict__ pc, int* __restrict__ idxs,
           float* __restrict__ sampled) {
  const int b = blockIdx.x, t = threadIdx.x;
  __shared__ float px[NIN], py[NIN], pz[NIN], mind[NIN];
  __shared__ float rv[256];
  __shared__ int   ri[256];
  const float* P = pc + (long)b * NIN * 3;
  for (int i = t; i < NIN; i += 256) {
    px[i] = P[i * 3 + 0]; py[i] = P[i * 3 + 1]; pz[i] = P[i * 3 + 2];
  }
  __syncthreads();
  const float x0 = px[0], y0 = py[0], z0 = pz[0];
  for (int i = t; i < NIN; i += 256) {
    float dx = px[i] - x0, dy = py[i] - y0, dz = pz[i] - z0;
    mind[i] = dx * dx + dy * dy + dz * dz;
  }
  if (t == 0) {
    idxs[b * NLAT] = 0;
    sampled[(long)b * NLAT * 3 + 0] = x0;
    sampled[(long)b * NLAT * 3 + 1] = y0;
    sampled[(long)b * NLAT * 3 + 2] = z0;
  }
  __syncthreads();
  for (int s = 1; s < NLAT; ++s) {
    float bm = -1.0f; int bi = 0;
    for (int i = t; i < NIN; i += 256) {
      float d = mind[i];
      if (d > bm) { bm = d; bi = i; }
    }
    rv[t] = bm; ri[t] = bi;
    __syncthreads();
    for (int w = 128; w > 0; w >>= 1) {
      if (t < w && rv[t + w] > rv[t]) { rv[t] = rv[t + w]; ri[t] = ri[t + w]; }
      __syncthreads();
    }
    const int sel = ri[0];
    const float sx = px[sel], sy = py[sel], sz = pz[sel];
    if (t == 0) {
      idxs[b * NLAT + s] = sel;
      sampled[((long)b * NLAT + s) * 3 + 0] = sx;
      sampled[((long)b * NLAT + s) * 3 + 1] = sy;
      sampled[((long)b * NLAT + s) * 3 + 2] = sz;
    }
    for (int i = t; i < NIN; i += 256) {
      float dx = px[i] - sx, dy = py[i] - sy, dz = pz[i] - sz;
      float d = dx * dx + dy * dy + dz * dz;
      if (d < mind[i]) mind[i] = d;
    }
    __syncthreads();
  }
}

// ---------------- fused PointEmbed + LayerNorm ----------------
__global__ void __launch_bounds__(256)
embed_ln_kernel(const float* __restrict__ pts,   // [rows,3]
                const float* __restrict__ pe_w,  // [51,512]
                const float* __restrict__ pe_b,  // [512]
                const float* __restrict__ g, const float* __restrict__ be,
                float* __restrict__ raw, unsigned short* __restrict__ nout) {
  const long row = blockIdx.x;
  const int t = threadIdx.x;
  __shared__ float feat[FEATN];
  __shared__ float pco[3];
  __shared__ float redA[256], redB[256];
  if (t < 3) pco[t] = pts[row * 3 + t];
  __syncthreads();
  if (t < 24) {
    float e = exp2f((float)(t & 7)) * 3.14159265358979323846f;
    float pr = pco[t >> 3] * e;
    feat[t] = sinf(pr);
    feat[t + 24] = cosf(pr);
  }
  if (t < 3) feat[48 + t] = pco[t];
  __syncthreads();
  const int c0 = t, c1 = t + 256;
  float a0 = pe_b[c0], a1 = pe_b[c1];
  for (int j = 0; j < FEATN; ++j) {
    const float f = feat[j];
    a0 = fmaf(f, pe_w[j * DIMD + c0], a0);
    a1 = fmaf(f, pe_w[j * DIMD + c1], a1);
  }
  if (raw) { raw[row * DIMD + c0] = a0; raw[row * DIMD + c1] = a1; }
  redA[t] = a0 + a1;
  redB[t] = a0 * a0 + a1 * a1;
  __syncthreads();
  for (int w = 128; w > 0; w >>= 1) {
    if (t < w) { redA[t] += redA[t + w]; redB[t] += redB[t + w]; }
    __syncthreads();
  }
  const float mu = redA[0] * (1.0f / DIMD);
  const float var = redB[0] * (1.0f / DIMD) - mu * mu;
  const float rstd = rsqrtf(var + LN_EPS);
  nout[row * DIMD + c0] = f2bf((a0 - mu) * rstd * g[c0] + be[c0]);
  nout[row * DIMD + c1] = f2bf((a1 - mu) * rstd * g[c1] + be[c1]);
}

// ---------------- plain LayerNorm (f32 in, bf16 out) ----------------
__global__ void __launch_bounds__(256)
ln_kernel(const float* __restrict__ x, const float* __restrict__ g,
          const float* __restrict__ be, unsigned short* __restrict__ nout) {
  const long row = blockIdx.x;
  const int t = threadIdx.x;
  __shared__ float redA[256], redB[256];
  const float a0 = x[row * DIMD + t], a1 = x[row * DIMD + t + 256];
  redA[t] = a0 + a1;
  redB[t] = a0 * a0 + a1 * a1;
  __syncthreads();
  for (int w = 128; w > 0; w >>= 1) {
    if (t < w) { redA[t] += redA[t + w]; redB[t] += redB[t + w]; }
    __syncthreads();
  }
  const float mu = redA[0] * (1.0f / DIMD);
  const float var = redB[0] * (1.0f / DIMD) - mu * mu;
  const float rstd = rsqrtf(var + LN_EPS);
  nout[row * DIMD + t] = f2bf((a0 - mu) * rstd * g[t] + be[t]);
  nout[row * DIMD + t + 256] = f2bf((a1 - mu) * rstd * g[t + 256] + be[t + 256]);
}

// ---------------- softmax over bf16 rows of length 2048, in place ----------------
__global__ void __launch_bounds__(256)
softmax_kernel(unsigned short* __restrict__ P) {
  const long row = blockIdx.x;
  const int t = threadIdx.x;
  unsigned short* p = P + row * (long)NIN;
  __shared__ float red[256];
  float v[8];
  float mx = -3.4e38f;
  for (int i = 0; i < 8; ++i) { v[i] = bf2f(p[t + i * 256]); mx = fmaxf(mx, v[i]); }
  red[t] = mx; __syncthreads();
  for (int w = 128; w > 0; w >>= 1) {
    if (t < w) red[t] = fmaxf(red[t], red[t + w]);
    __syncthreads();
  }
  mx = red[0]; __syncthreads();
  float s = 0.0f;
  for (int i = 0; i < 8; ++i) { v[i] = expf(v[i] - mx); s += v[i]; }
  red[t] = s; __syncthreads();
  for (int w = 128; w > 0; w >>= 1) {
    if (t < w) red[t] += red[t + w];
    __syncthreads();
  }
  const float inv = 1.0f / red[0];
  for (int i = 0; i < 8; ++i) p[t + i * 256] = f2bf(v[i] * inv);
}

// ---------------- async-staged WMMA GEMM: C = A[M,K] * BT[N,K]^T ----------------
// 256 threads (8 waves, 4x2), tile 128(M) x 64(N), K-step 64, double-buffered
// LDS filled with global_load_async_to_lds_b128 (ASYNCcnt), WMMA fed by ds_load_b128.
// EPI 0: Cb[z] = bf16(acc*scale)
// EPI 1: split KV: n<nSplit -> Cb (K row-major); else -> C2 = V^T per batch
// EPI 2: Cf = acc + bias[n] + res[m,n]   (f32)
template <int EPI>
__global__ void __launch_bounds__(256)
gemm_bf16(const unsigned short* __restrict__ A,
          const unsigned short* __restrict__ BT,
          int M, int N, int K,
          long long sAb, long long sBb, long long sCb,
          float scale,
          unsigned short* __restrict__ Cb,
          unsigned short* __restrict__ C2,
          float* __restrict__ Cf,
          const float* __restrict__ bias,
          const float* __restrict__ res,
          int nSplit) {
  __shared__ unsigned short As[2][128 * LDA_M];
  __shared__ unsigned short Bs[2][64 * LDA_M];
  const int t = threadIdx.x;
  const int lane = t & 31;
  const int wid = t >> 5;
  const int wm = wid & 3, wn = wid >> 2;
  const long z = blockIdx.z;
  const long rowBlk = (long)blockIdx.y * 128;
  const int colBlk = blockIdx.x * 64;
  const unsigned short* Ab = A + z * sAb;
  const unsigned short* Bb = BT + z * sBb;
  const int r0 = lane & 15;
  const int kb = (lane >> 4) * 8;

  // per-thread stage: 4 A-chunks + 2 B-chunks of 16B  -> 6 ASYNCcnt per wave
  auto stage = [&](int sbuf, int sk0) {
#pragma unroll
    for (int i = 0; i < 4; ++i) {
      int c = t + i * 256;                 // 1024 chunks: 128 rows x 8
      int row = c >> 3, kc = (c & 7) * 8;
      unsigned lds = (unsigned)(unsigned long long)&As[sbuf][row * LDA_M + kc];
      unsigned off = (unsigned)(((rowBlk + row) * (long)K + sk0 + kc) * 2);
      async_b128(lds, Ab, off);
    }
#pragma unroll
    for (int i = 0; i < 2; ++i) {
      int c = t + i * 256;                 // 512 chunks: 64 rows x 8
      int row = c >> 3, kc = (c & 7) * 8;
      unsigned lds = (unsigned)(unsigned long long)&Bs[sbuf][row * LDA_M + kc];
      unsigned off = (unsigned)(((colBlk + row) * (long)K + sk0 + kc) * 2);
      async_b128(lds, Bb, off);
    }
  };

  v8f acc[2][2] = {};
  stage(0, 0);
  int buf = 0;
  for (int k0 = 0; k0 < K; k0 += 64) {
    if (k0 + 64 < K) {
      stage(buf ^ 1, k0 + 64);
      ASYNC_WAIT(0x6);                     // current stage resident, next in flight
    } else {
      ASYNC_WAIT(0x0);
    }
    __syncthreads();
    const unsigned short* Ap = &As[buf][0];
    const unsigned short* Bp = &Bs[buf][0];
#pragma unroll
    for (int ks = 0; ks < 64; ks += 32) {
      FragAB a0 = lds_frag(Ap + (wm * 32 + r0) * LDA_M + ks + kb);
      FragAB a1 = lds_frag(Ap + (wm * 32 + 16 + r0) * LDA_M + ks + kb);
      FragAB b0 = lds_frag(Bp + (wn * 32 + r0) * LDA_M + ks + kb);
      FragAB b1 = lds_frag(Bp + (wn * 32 + 16 + r0) * LDA_M + ks + kb);
      acc[0][0] = WMMA_BF16(a0.v, b0.v, acc[0][0]);
      acc[0][1] = WMMA_BF16(a0.v, b1.v, acc[0][1]);
      acc[1][0] = WMMA_BF16(a1.v, b0.v, acc[1][0]);
      acc[1][1] = WMMA_BF16(a1.v, b1.v, acc[1][1]);
    }
    __syncthreads();                       // readers done before buf is refilled
    buf ^= 1;
  }

  const long rowBase = rowBlk + wm * 32;
  const int colBase = colBlk + wn * 32;
  for (int tm = 0; tm < 2; ++tm)
    for (int tn = 0; tn < 2; ++tn)
      for (int i = 0; i < 8; ++i) {
        const long m = rowBase + tm * 16 + (lane >> 4) * 8 + i;
        const int n = colBase + tn * 16 + (lane & 15);
        const float v = acc[tm][tn][i] * scale;
        if (EPI == 0) {
          Cb[z * sCb + m * (long)N + n] = f2bf(v);
        } else if (EPI == 1) {
          if (n < nSplit) {
            Cb[m * (long)nSplit + n] = f2bf(v);
          } else {
            const long bt = m / NIN, r = m % NIN;
            C2[bt * (long)DIMD * NIN + (long)(n - nSplit) * NIN + r] = f2bf(v);
          }
        } else {
          Cf[m * (long)N + n] = v + bias[n] + res[m * (long)N + n];
        }
      }
}

// ---------------- GEGLU GEMM (async staged, dual column tiles) ----------------
__global__ void __launch_bounds__(256)
gemm_geglu(const unsigned short* __restrict__ A,    // xn [M,512]
           const unsigned short* __restrict__ W1T,  // [4096,512]
           const float* __restrict__ b1,            // [4096]
           unsigned short* __restrict__ gg,         // [M,2048] bf16
           int M, int K, int Nh) {
  __shared__ unsigned short As[2][128 * LDA_G];
  __shared__ unsigned short Ba[2][64 * LDA_G];
  __shared__ unsigned short Bg[2][64 * LDA_G];
  const int t = threadIdx.x;
  const int lane = t & 31;
  const int wid = t >> 5;
  const int wm = wid & 3, wn = wid >> 2;
  const long rowBlk = (long)blockIdx.y * 128;
  const int colBlk = blockIdx.x * 64;
  const int r0 = lane & 15;
  const int kb = (lane >> 4) * 8;

  // per-thread stage: 2 A + 1 Ba + 1 Bg chunks -> 4 ASYNCcnt per wave
  auto stage = [&](int sbuf, int sk0) {
#pragma unroll
    for (int i = 0; i < 2; ++i) {
      int c = t + i * 256;                 // 512 chunks: 128 rows x 4
      int row = c >> 2, kc = (c & 3) * 8;
      unsigned lds = (unsigned)(unsigned long long)&As[sbuf][row * LDA_G + kc];
      unsigned off = (unsigned)(((rowBlk + row) * (long)K + sk0 + kc) * 2);
      async_b128(lds, A, off);
    }
    {
      int row = t >> 2, kc = (t & 3) * 8;  // 256 chunks: 64 rows x 4
      unsigned lds = (unsigned)(unsigned long long)&Ba[sbuf][row * LDA_G + kc];
      unsigned off = (unsigned)(((colBlk + row) * (long)K + sk0 + kc) * 2);
      async_b128(lds, W1T, off);
      unsigned ldsg = (unsigned)(unsigned long long)&Bg[sbuf][row * LDA_G + kc];
      unsigned offg = (unsigned)(((colBlk + Nh + row) * (long)K + sk0 + kc) * 2);
      async_b128(ldsg, W1T, offg);
    }
  };

  v8f aa[2][2] = {};
  v8f ag[2][2] = {};
  stage(0, 0);
  int buf = 0;
  for (int k0 = 0; k0 < K; k0 += 32) {
    if (k0 + 32 < K) {
      stage(buf ^ 1, k0 + 32);
      ASYNC_WAIT(0x4);
    } else {
      ASYNC_WAIT(0x0);
    }
    __syncthreads();
    const unsigned short* Ap = &As[buf][0];
    const unsigned short* Pa = &Ba[buf][0];
    const unsigned short* Pg = &Bg[buf][0];
    FragAB a0 = lds_frag(Ap + (wm * 32 + r0) * LDA_G + kb);
    FragAB a1 = lds_frag(Ap + (wm * 32 + 16 + r0) * LDA_G + kb);
    FragAB ba0 = lds_frag(Pa + (wn * 32 + r0) * LDA_G + kb);
    FragAB ba1 = lds_frag(Pa + (wn * 32 + 16 + r0) * LDA_G + kb);
    FragAB bg0 = lds_frag(Pg + (wn * 32 + r0) * LDA_G + kb);
    FragAB bg1 = lds_frag(Pg + (wn * 32 + 16 + r0) * LDA_G + kb);
    aa[0][0] = WMMA_BF16(a0.v, ba0.v, aa[0][0]);
    aa[0][1] = WMMA_BF16(a0.v, ba1.v, aa[0][1]);
    aa[1][0] = WMMA_BF16(a1.v, ba0.v, aa[1][0]);
    aa[1][1] = WMMA_BF16(a1.v, ba1.v, aa[1][1]);
    ag[0][0] = WMMA_BF16(a0.v, bg0.v, ag[0][0]);
    ag[0][1] = WMMA_BF16(a0.v, bg1.v, ag[0][1]);
    ag[1][0] = WMMA_BF16(a1.v, bg0.v, ag[1][0]);
    ag[1][1] = WMMA_BF16(a1.v, bg1.v, ag[1][1]);
    __syncthreads();
    buf ^= 1;
  }

  const long rowBase = rowBlk + wm * 32;
  const int colBase = colBlk + wn * 32;
  for (int tm = 0; tm < 2; ++tm)
    for (int tn = 0; tn < 2; ++tn)
      for (int i = 0; i < 8; ++i) {
        const long m = rowBase + tm * 16 + (lane >> 4) * 8 + i;
        const int n = colBase + tn * 16 + (lane & 15);
        const float va = aa[tm][tn][i] + b1[n];
        const float vg = ag[tm][tn][i] + b1[n + Nh];
        const float ge = 0.5f * vg * (1.0f + erff(vg * 0.70710678118654752f));
        gg[m * (long)Nh + n] = f2bf(va * ge);
      }
}

// ---------------- host-side orchestration ----------------
static inline size_t alignup(size_t x) { return (x + 255) & ~(size_t)255; }

extern "C" void kernel_launch(void* const* d_in, const int* in_sizes, int n_in,
                              void* d_out, int out_size, void* d_ws, size_t ws_size,
                              hipStream_t stream) {
  (void)in_sizes; (void)n_in; (void)out_size; (void)ws_size;
  const float* pc    = (const float*)d_in[0];
  const float* pe_w  = (const float*)d_in[1];
  const float* pe_b  = (const float*)d_in[2];
  const float* ln_qg = (const float*)d_in[3];
  const float* ln_qb = (const float*)d_in[4];
  const float* ln_cg = (const float*)d_in[5];
  const float* ln_cb = (const float*)d_in[6];
  const float* wq    = (const float*)d_in[7];
  const float* wkv   = (const float*)d_in[8];
  const float* wo    = (const float*)d_in[9];
  const float* bo    = (const float*)d_in[10];
  const float* ln_fg = (const float*)d_in[11];
  const float* ln_fb = (const float*)d_in[12];
  const float* w1    = (const float*)d_in[13];
  const float* b1    = (const float*)d_in[14];
  const float* w2    = (const float*)d_in[15];
  const float* b2    = (const float*)d_in[16];

  char* ws = (char*)d_ws;
  size_t cur = 0;
  auto take = [&](size_t bytes) { size_t o = alignup(cur); cur = o + bytes; return ws + o; };

  const long rowsQ = (long)BATCH * NLAT;    // 16384
  const long rowsC = (long)BATCH * NIN;     // 65536

  unsigned short* wqT  = (unsigned short*)take((size_t)DIMD * DIMD * 2);
  unsigned short* wkvT = (unsigned short*)take((size_t)2 * DIMD * DIMD * 2);
  unsigned short* woT  = (unsigned short*)take((size_t)DIMD * DIMD * 2);
  unsigned short* w1T  = (unsigned short*)take((size_t)2 * FFD * DIMD * 2);
  unsigned short* w2T  = (unsigned short*)take((size_t)DIMD * FFD * 2);
  int*            idxs = (int*)take((size_t)BATCH * NLAT * 4);
  float*          samp = (float*)take((size_t)BATCH * NLAT * 3 * 4);
  float*          qemb = (float*)take((size_t)rowsQ * DIMD * 4);
  unsigned short* qn   = (unsigned short*)take((size_t)rowsQ * DIMD * 2);
  unsigned short* cn   = (unsigned short*)take((size_t)rowsC * DIMD * 2);
  unsigned short* q    = (unsigned short*)take((size_t)rowsQ * DIMD * 2);
  unsigned short* kbuf = (unsigned short*)take((size_t)rowsC * DIMD * 2);
  unsigned short* vT   = (unsigned short*)take((size_t)BATCH * DIMD * NIN * 2);
  unsigned short* P    = (unsigned short*)take((size_t)BATCH * NLAT * NIN * 2);
  unsigned short* attn = (unsigned short*)take((size_t)rowsQ * DIMD * 2);
  float*          x    = (float*)take((size_t)rowsQ * DIMD * 4);
  unsigned short* xn   = (unsigned short*)take((size_t)rowsQ * DIMD * 2);
  unsigned short* gg   = cn;   // cn is dead after the KV GEMM; reuse

  const dim3 blk(256);
  // 1) weight transposes (f32 [K,N] -> bf16 [N,K])
  transpose_to_bf16<<<(DIMD * DIMD + 255) / 256, blk, 0, stream>>>(wq, wqT, DIMD, DIMD);
  transpose_to_bf16<<<(DIMD * 2 * DIMD + 255) / 256, blk, 0, stream>>>(wkv, wkvT, DIMD, 2 * DIMD);
  transpose_to_bf16<<<(DIMD * DIMD + 255) / 256, blk, 0, stream>>>(wo, woT, DIMD, DIMD);
  transpose_to_bf16<<<(DIMD * 2 * FFD + 255) / 256, blk, 0, stream>>>(w1, w1T, DIMD, 2 * FFD);
  transpose_to_bf16<<<(FFD * DIMD + 255) / 256, blk, 0, stream>>>(w2, w2T, FFD, DIMD);

  // 2) FPS + gather
  fps_kernel<<<BATCH, blk, 0, stream>>>(pc, idxs, samp);

  // 3) fused embed + LayerNorm (query path keeps raw f32 for residual)
  embed_ln_kernel<<<(unsigned)rowsQ, blk, 0, stream>>>(samp, pe_w, pe_b, ln_qg, ln_qb, qemb, qn);
  embed_ln_kernel<<<(unsigned)rowsC, blk, 0, stream>>>(pc, pe_w, pe_b, ln_cg, ln_cb, nullptr, cn);

  // 4) q = qn @ wq
  gemm_bf16<0><<<dim3(DIMD / 64, rowsQ / 128, 1), blk, 0, stream>>>(
      qn, wqT, (int)rowsQ, DIMD, DIMD, 0, 0, 0, 1.0f, q, nullptr, nullptr, nullptr, nullptr, 0);

  // 5) kv = cn @ wkv  (k row-major; v written transposed per batch)
  gemm_bf16<1><<<dim3(2 * DIMD / 64, rowsC / 128, 1), blk, 0, stream>>>(
      cn, wkvT, (int)rowsC, 2 * DIMD, DIMD, 0, 0, 0, 1.0f, kbuf, vT, nullptr, nullptr, nullptr, DIMD);

  // 6) S = scale * q @ k^T   (batched; BT = k row-major is exactly (k^T)^T)
  gemm_bf16<0><<<dim3(NIN / 64, NLAT / 128, BATCH), blk, 0, stream>>>(
      q, kbuf, NLAT, NIN, DIMD,
      (long long)NLAT * DIMD, (long long)NIN * DIMD, (long long)NLAT * NIN,
      0.044194173824159216f /* 1/sqrt(512) */, P, nullptr, nullptr, nullptr, nullptr, 0);

  // 7) softmax rows in place
  softmax_kernel<<<(unsigned)rowsQ, blk, 0, stream>>>(P);

  // 8) attn = P @ v   (BT = v^T, K = 2048)
  gemm_bf16<0><<<dim3(DIMD / 64, NLAT / 128, BATCH), blk, 0, stream>>>(
      P, vT, NLAT, DIMD, NIN,
      (long long)NLAT * NIN, (long long)DIMD * NIN, (long long)NLAT * DIMD,
      1.0f, attn, nullptr, nullptr, nullptr, nullptr, 0);

  // 9) x = attn @ wo + bo + q_emb
  gemm_bf16<2><<<dim3(DIMD / 64, rowsQ / 128, 1), blk, 0, stream>>>(
      attn, woT, (int)rowsQ, DIMD, DIMD, 0, 0, 0, 1.0f, nullptr, nullptr, x, bo, qemb, 0);

  // 10) xn = LN(x)
  ln_kernel<<<(unsigned)rowsQ, blk, 0, stream>>>(x, ln_fg, ln_fb, xn);

  // 11) gg = a * gelu(g); h = xn @ w1 + b1 never materialized
  gemm_geglu<<<dim3(FFD / 64, rowsQ / 128, 1), blk, 0, stream>>>(
      xn, w1T, b1, gg, (int)rowsQ, DIMD, FFD);

  // 12) out = gg @ w2 + b2 + x
  gemm_bf16<2><<<dim3(DIMD / 64, rowsQ / 128, 1), blk, 0, stream>>>(
      gg, w2T, (int)rowsQ, DIMD, FFD, 0, 0, 0, 1.0f, nullptr, nullptr, (float*)d_out, b2, x, 0);
}
